// RGLRU_42477226557876
// MI455X (gfx1250) — matrix operand order repaired and checked by reading, not compiled, not run on hardware
//
#include <hip/hip_runtime.h>
#include <hip/hip_bf16.h>
#include <stdint.h>
#include <stddef.h>

// RG-LRU for MI455X (gfx1250): bf16 WMMA dual-gate GEMM + chunked parallel scan.
// Round 2: bf16-converted x (A traffic halved, cvt out of hot loop),
// 32x32x(2-gate) macro-tile per wave (8 WMMA accumulators, 4x better reuse),
// v_rcp_f32-based sigmoid epilogue.

typedef __attribute__((ext_vector_type(16))) __bf16 v16bf;
typedef __attribute__((ext_vector_type(8)))  __bf16 v8bf;
typedef __attribute__((ext_vector_type(8)))  float  v8f;
typedef __attribute__((ext_vector_type(8)))  float  f32x8;

#define DIMK 1024          // DIM == HID
#define BATB 8
#define SEQT 4096
#define MTOT (BATB*SEQT)   // 32768 rows
#define NCH  32            // scan chunks per sequence
#define LCH  (SEQT/NCH)    // 128 steps per chunk

__device__ __forceinline__ float sigf(float v) {
    // gate-accuracy sigmoid: single v_exp + v_rcp, no IEEE divide chain
    return __builtin_amdgcn_rcpf(1.0f + __expf(-v));
}

// ---------------------------------------------------------------------------
// Pass 0a: x -> bf16 (64 MiB), 8 elements/thread, coalesced.
// ---------------------------------------------------------------------------
__global__ void prep_x_kernel(const float* __restrict__ x, __bf16* __restrict__ xb) {
    size_t i = ((size_t)blockIdx.x * blockDim.x + threadIdx.x) * 8;
    f32x8 v = *(const f32x8*)(x + i);
    v8bf o;
#pragma unroll
    for (int e = 0; e < 8; ++e) o[e] = (__bf16)v[e];
    *(v8bf*)(xb + i) = o;
}

// ---------------------------------------------------------------------------
// Pass 0b: Wt[h][d] = bf16(W[d][h]) for both weights (B-fragments become
// contiguous 32B loads per lane). 4 MiB total; cost negligible.
// ---------------------------------------------------------------------------
__global__ void prep_w_kernel(const float* __restrict__ Wa, const float* __restrict__ Wx,
                              __bf16* __restrict__ Wta, __bf16* __restrict__ Wtx) {
    int idx = blockIdx.x * blockDim.x + threadIdx.x;   // over 1024*1024
    int h = idx >> 10;
    int d = idx & (DIMK - 1);
    Wta[idx] = (__bf16)Wa[(size_t)d * DIMK + h];
    Wtx[idx] = (__bf16)Wx[(size_t)d * DIMK + h];
}

// A-fragment (16-bit 16x32 layout): lanes 0-15 = M row, kh = lane>>4.
// elems 0..15 = K kbase+kh*16.. (contiguous, matches B) -> two 16B loads.
__device__ __forceinline__ v16bf load_afrag(const __bf16* rowp, int k, int kh) {
    v8bf lo = *(const v8bf*)(rowp + k + kh * 16);
    v8bf hi = *(const v8bf*)(rowp + k + kh * 16 + 8);
    v16bf a;
#pragma unroll
    for (int e = 0; e < 8; ++e) { a[e] = lo[e]; a[8 + e] = hi[e]; }
    return a;
}

// ---------------------------------------------------------------------------
// Pass 1: fused dual-gate GEMM. One wave owns a 32(M)x32(N) block of BOTH
// gates: accR[mi][ni], accI[mi][ni] (8 v8f accumulators). Per 32-k step:
// 2 A fragments + 4 B fragments feed 8 v_wmma_f32_16x16x32_bf16.
//   B layout (16-bit 32x16): lane&15 = N column; elems 0..15 = K kbase+kh*16..
//   C/D layout: lane&15 = N; VGPR j = row j + 8*(lane>=16)
// ---------------------------------------------------------------------------
__global__ void gates_wmma_kernel(const __bf16* __restrict__ xb,
                                  const float*  __restrict__ x,
                                  const __bf16* __restrict__ Wta,
                                  const __bf16* __restrict__ Wtx,
                                  const float*  __restrict__ ba,
                                  const float*  __restrict__ bx,
                                  const float*  __restrict__ Lam,
                                  float* __restrict__ at_out,
                                  float* __restrict__ bt_out) {
    const int lane  = threadIdx.x & 31;
    const int wave  = threadIdx.x >> 5;
    const int mpair = blockIdx.x * 8 + wave;   // 1024 pairs of M-tiles
    const int npair = blockIdx.y;              // 32 pairs of N-tiles
    const int m0 = mpair * 32;
    const int n0 = npair * 32;
    const int row = lane & 15;
    const int kh  = lane >> 4;

    const __bf16* a0p = xb + (size_t)(m0 + row) * DIMK;
    const __bf16* a1p = a0p + (size_t)16 * DIMK;
    const __bf16* bA0 = Wta + (size_t)(n0 + row) * DIMK + kh * 16;
    const __bf16* bA1 = bA0 + (size_t)16 * DIMK;
    const __bf16* bX0 = Wtx + (size_t)(n0 + row) * DIMK + kh * 16;
    const __bf16* bX1 = bX0 + (size_t)16 * DIMK;

    v8f accR[2][2] = {};
    v8f accI[2][2] = {};
    for (int k = 0; k < DIMK; k += 32) {
        v16bf a0 = load_afrag(a0p, k, kh);
        v16bf a1 = load_afrag(a1p, k, kh);
        v16bf b0 = *(const v16bf*)(bA0 + k);
        v16bf b1 = *(const v16bf*)(bA1 + k);
        v16bf c0 = *(const v16bf*)(bX0 + k);
        v16bf c1 = *(const v16bf*)(bX1 + k);
        accR[0][0] = __builtin_amdgcn_wmma_f32_16x16x32_bf16(false, a0, false, b0, (short)0, accR[0][0], false, false);
        accR[0][1] = __builtin_amdgcn_wmma_f32_16x16x32_bf16(false, a0, false, b1, (short)0, accR[0][1], false, false);
        accR[1][0] = __builtin_amdgcn_wmma_f32_16x16x32_bf16(false, a1, false, b0, (short)0, accR[1][0], false, false);
        accR[1][1] = __builtin_amdgcn_wmma_f32_16x16x32_bf16(false, a1, false, b1, (short)0, accR[1][1], false, false);
        accI[0][0] = __builtin_amdgcn_wmma_f32_16x16x32_bf16(false, a0, false, c0, (short)0, accI[0][0], false, false);
        accI[0][1] = __builtin_amdgcn_wmma_f32_16x16x32_bf16(false, a0, false, c1, (short)0, accI[0][1], false, false);
        accI[1][0] = __builtin_amdgcn_wmma_f32_16x16x32_bf16(false, a1, false, c0, (short)0, accI[1][0], false, false);
        accI[1][1] = __builtin_amdgcn_wmma_f32_16x16x32_bf16(false, a1, false, c1, (short)0, accI[1][1], false, false);
    }

#pragma unroll
    for (int ni = 0; ni < 2; ++ni) {
        const int h = n0 + ni * 16 + row;      // this lane's output column
        const float bav = ba[h];
        const float bxv = bx[h];
        const float aST = sigf(Lam[h]);        // sigmoid(Lambda)
#pragma unroll
        for (int mi = 0; mi < 2; ++mi) {
#pragma unroll
            for (int j = 0; j < 8; ++j) {
                const int m = m0 + mi * 16 + j + kh * 8;   // C/D row for VGPR j
                float r  = sigf(accR[mi][ni][j] + bav);
                float ig = sigf(accI[mi][ni][j] + bxv);
                float at = aST * exp2f(-3.0f * r);         // a * 8^{-r}
                float xv = x[(size_t)m * DIMK + h];
                float bt = sqrtf(fmaxf(1.0f - at * at, 0.0f)) * ig * xv;
                at_out[(size_t)m * DIMK + h] = at;
                bt_out[(size_t)m * DIMK + h] = bt;         // bt staged in d_out
            }
        }
    }
}

// ---------------------------------------------------------------------------
// Pass 2a: per-chunk carries. state (A, B): absorbing step t -> A*=a_t, B=a_t*B+b_t.
// Thread id = ((b*NCH)+c)*DIMK + h -> adjacent threads coalesce over h.
// ---------------------------------------------------------------------------
__global__ void scan_carry_kernel(const float* __restrict__ at, const float* __restrict__ bt,
                                  float* __restrict__ cA, float* __restrict__ cB) {
    int tid = blockIdx.x * blockDim.x + threadIdx.x;  // BATB*NCH*DIMK = 262144
    int h = tid & (DIMK - 1);
    int c = (tid >> 10) & (NCH - 1);
    int b = tid >> 15;
    size_t base = ((size_t)b * SEQT + (size_t)c * LCH) * DIMK + h;
    float A = 1.0f, Bs = 0.0f;
    for (int t = 0; t < LCH; ++t) {
        float a = at[base + (size_t)t * DIMK];
        float v = bt[base + (size_t)t * DIMK];
        Bs = fmaf(a, Bs, v);
        A *= a;
    }
    cA[tid] = A;
    cB[tid] = Bs;
}

// Pass 2b: sequential prefix over the 32 chunk-carries per (b,h). 8192 threads, tiny.
__global__ void scan_prefix_kernel(const float* __restrict__ cA, const float* __restrict__ cB,
                                   float* __restrict__ Hs) {
    int tid = blockIdx.x * blockDim.x + threadIdx.x;  // BATB*DIMK = 8192
    int h = tid & (DIMK - 1);
    int b = tid >> 10;
    float hp = 0.0f;
    for (int c = 0; c < NCH; ++c) {
        int i = (b * NCH + c) * DIMK + h;
        Hs[i] = hp;                    // hidden state at chunk start
        hp = fmaf(cA[i], hp, cB[i]);
    }
}

// Pass 2c: apply — rescan each chunk from its incoming state; overwrite bt with y in place.
__global__ void scan_apply_kernel(const float* __restrict__ at, const float* __restrict__ Hs,
                                  float* __restrict__ y) {
    int tid = blockIdx.x * blockDim.x + threadIdx.x;
    int h = tid & (DIMK - 1);
    int c = (tid >> 10) & (NCH - 1);
    int b = tid >> 15;
    size_t base = ((size_t)b * SEQT + (size_t)c * LCH) * DIMK + h;
    float hc = Hs[tid];
    for (int t = 0; t < LCH; ++t) {
        size_t idx = base + (size_t)t * DIMK;
        hc = fmaf(at[idx], hc, y[idx]);   // y currently holds bt
        y[idx] = hc;
    }
}

// ---------------------------------------------------------------------------
extern "C" void kernel_launch(void* const* d_in, const int* in_sizes, int n_in,
                              void* d_out, int out_size, void* d_ws, size_t ws_size,
                              hipStream_t stream) {
    (void)in_sizes; (void)n_in; (void)out_size; (void)ws_size;
    const float* x   = (const float*)d_in[0];
    const float* Wa  = (const float*)d_in[1];
    const float* Wx  = (const float*)d_in[2];
    const float* ba  = (const float*)d_in[3];
    const float* bx  = (const float*)d_in[4];
    const float* Lam = (const float*)d_in[5];
    float* out = (float*)d_out;                 // bt staging, then y in place

    // Workspace layout
    char* w = (char*)d_ws;
    __bf16* xb  = (__bf16*)w;                                   //  64 MiB
    size_t  off = (size_t)MTOT * DIMK * 2;
    __bf16* Wta = (__bf16*)(w + off);                           //   2 MiB
    __bf16* Wtx = (__bf16*)(w + off + ((size_t)1 << 21));       //   2 MiB
    float*  atb = (float*)(w + off + ((size_t)2 << 21));        // 128 MiB
    float*  cA  = (float*)(w + off + ((size_t)2 << 21) + (size_t)MTOT * DIMK * 4);
    float*  cB  = cA + (size_t)BATB * DIMK * NCH;               //   1 MiB each
    float*  Hs  = cB + (size_t)BATB * DIMK * NCH;

    // Pass 0: bf16 casts (x) and weight transpose+cast
    prep_x_kernel<<<((size_t)MTOT * DIMK / 8) / 256, 256, 0, stream>>>(x, xb);
    prep_w_kernel<<<(DIMK * DIMK) / 256, 256, 0, stream>>>(Wa, Wx, Wta, Wtx);

    // Pass 1: dual-gate bf16 WMMA GEMM + elementwise -> at (ws), bt (d_out)
    dim3 ggrid((MTOT / 32) / 8, DIMK / 32);     // (128, 32), 8 waves/block
    gates_wmma_kernel<<<ggrid, 256, 0, stream>>>(xb, x, Wta, Wtx, ba, bx, Lam, atb, out);

    // Pass 2: chunked linear-recurrence scan along T
    scan_carry_kernel<<<(BATB * NCH * DIMK) / 256, 256, 0, stream>>>(atb, out, cA, cB);
    scan_prefix_kernel<<<(BATB * DIMK) / 256, 256, 0, stream>>>(cA, cB, Hs);
    scan_apply_kernel<<<(BATB * NCH * DIMK) / 256, 256, 0, stream>>>(atb, Hs, out);
}
